// FlashAttention_798863917297
// MI455X (gfx1250) — matrix operand order repaired and checked by reading, not compiled, not run on hardware
//
#include <hip/hip_runtime.h>
#include <hip/hip_bf16.h>

// ---------------- problem constants ----------------
#define DIMM   1024
#define NHEAD  16
#define HD     64
#define SEQ    2048
#define BATCH  2
#define NTOK   (BATCH * SEQ)          // 4096
#define QK_SCALE 0.125f               // 64^-0.5

typedef __attribute__((ext_vector_type(16))) __bf16 v16bf;
typedef __attribute__((ext_vector_type(8)))  __bf16 v8bf;
typedef __attribute__((ext_vector_type(8)))  float  v8f;

// ---------------------------------------------------------------------------
// WMMA fragment loaders (wave32, v_wmma_f32_16x16x32_bf16 layouts, ISA 7.12.2)
//
// A (16x32, row-major source, leading dim lda):
//   lanes 0-15 : row = row0+lane, elems 0..7 = K k0..k0+7, elems 8..15 = k0+16..k0+23
//   lanes16-31 : row = row0+lane-16, K offsets +8 / +24
// B (32x16) built from an N-major ("W^T") source W[N][ldw]:
//   lanes 0-15 : col N = n0+lane,   elems 0..7 = W[n][k0..k0+7], elems 8..15 = W[n][k0+16..k0+23]
//   lanes16-31 : col N = n0+lane-16, K offsets +8 / +24
// ---------------------------------------------------------------------------
__device__ __forceinline__ v16bf load_frag_a(const __bf16* __restrict__ base,
                                             int lda, int row0, int k0) {
  const int lane = threadIdx.x & 31;
  const int half = lane >> 4;
  const __bf16* p = base + (row0 + (lane & 15)) * lda + k0 + half * 8;
  v8bf lo = *(const v8bf*)(p);
  v8bf hi = *(const v8bf*)(p + 16);
  v16bf out;
#pragma unroll
  for (int i = 0; i < 8; ++i) { out[i] = lo[i]; out[i + 8] = hi[i]; }
  return out;
}

__device__ __forceinline__ v16bf load_frag_bt(const __bf16* __restrict__ W,
                                              int ldw, int n0, int k0) {
  const int lane = threadIdx.x & 31;
  const int half = lane >> 4;
  const __bf16* p = W + (n0 + (lane & 15)) * ldw + k0 + half * 8;
  v8bf lo = *(const v8bf*)(p);
  v8bf hi = *(const v8bf*)(p + 16);
  v16bf out;
#pragma unroll
  for (int i = 0; i < 8; ++i) { out[i] = lo[i]; out[i + 8] = hi[i]; }
  return out;
}

#define WMMA_BF16(A, B, C) \
  __builtin_amdgcn_wmma_f32_16x16x32_bf16(false, (A), false, (B), (short)0, (C), false, false)

// ---------------------------------------------------------------------------
// Kernel 0: fp32 -> bf16 convert
// ---------------------------------------------------------------------------
__global__ void cvt_f32_bf16(const float* __restrict__ in, __bf16* __restrict__ out, int n) {
  int i = blockIdx.x * blockDim.x + threadIdx.x;
  int stride = gridDim.x * blockDim.x;
  for (; i < n; i += stride) out[i] = (__bf16)in[i];
}

// ---------------------------------------------------------------------------
// Software-pipelined 16(M) x 64(N) wave-tile GEMM body: C = A_rm @ W^T.
// k-loop unrolled by 2 with role-alternating buffers: each buffer is reloaded
// only AFTER the WMMAs that consume it, so no cross-backedge register copies
// (no v_mov rotation storm), while one load batch is always in flight.
// Requires K % 64 == 0.
// ---------------------------------------------------------------------------
__device__ __forceinline__ void gemm_tile_16x64(const __bf16* __restrict__ A,
                                                const __bf16* __restrict__ W,
                                                int K, int m0, int n0, v8f acc[4]) {
  v16bf a0 = load_frag_a(A, K, m0, 0);
  v16bf b0[4];
#pragma unroll
  for (int t = 0; t < 4; ++t) b0[t] = load_frag_bt(W, K, n0 + t * 16, 0);

  int k0 = 0;
  for (; k0 + 64 < K; k0 += 64) {
    // stage 1: issue loads for k0+32, consume k0
    v16bf a1 = load_frag_a(A, K, m0, k0 + 32);
    v16bf b1[4];
#pragma unroll
    for (int t = 0; t < 4; ++t) b1[t] = load_frag_bt(W, K, n0 + t * 16, k0 + 32);
#pragma unroll
    for (int t = 0; t < 4; ++t) acc[t] = WMMA_BF16(a0, b0[t], acc[t]);
    // stage 2: reload the k0 buffers with k0+64, consume k0+32
    a0 = load_frag_a(A, K, m0, k0 + 64);
#pragma unroll
    for (int t = 0; t < 4; ++t) b0[t] = load_frag_bt(W, K, n0 + t * 16, k0 + 64);
#pragma unroll
    for (int t = 0; t < 4; ++t) acc[t] = WMMA_BF16(a1, b1[t], acc[t]);
  }
  // tail: k0 == K-64; frags for K-64 already resident, load K-32 and finish.
  {
    v16bf a1 = load_frag_a(A, K, m0, k0 + 32);
    v16bf b1[4];
#pragma unroll
    for (int t = 0; t < 4; ++t) b1[t] = load_frag_bt(W, K, n0 + t * 16, k0 + 32);
#pragma unroll
    for (int t = 0; t < 4; ++t) acc[t] = WMMA_BF16(a0, b0[t], acc[t]);
#pragma unroll
    for (int t = 0; t < 4; ++t) acc[t] = WMMA_BF16(a1, b1[t], acc[t]);
  }
}

// ---------------------------------------------------------------------------
// Kernel 1: qkv = x @ w_qkv^T   (M=4096, N=3072, K=1024).
// Epilogue scatters Q,K -> [B,H,L,hd]; V -> transposed [B,H,hd,L].
// Block = 8 waves; grid (256, 6).
// ---------------------------------------------------------------------------
__global__ __launch_bounds__(256) void qkv_gemm(const __bf16* __restrict__ X,
                                                const __bf16* __restrict__ W,
                                                __bf16* __restrict__ q_buf,
                                                __bf16* __restrict__ k_buf,
                                                __bf16* __restrict__ vt_buf) {
  const int lane = threadIdx.x & 31;
  const int wid  = threadIdx.x >> 5;
  const int half = lane >> 4;
  const int m0 = blockIdx.x * 16;
  const int n0 = blockIdx.y * 512 + wid * 64;

  v8f acc[4] = {};
  gemm_tile_16x64(X, W, DIMM, m0, n0, acc);

#pragma unroll
  for (int t = 0; t < 4; ++t)
#pragma unroll
    for (int j = 0; j < 8; ++j) {
      const int m = m0 + j + half * 8;          // token index
      const int n = n0 + t * 16 + (lane & 15);  // output feature in [0,3072)
      const __bf16 v = (__bf16)acc[t][j];
      const int b   = m >> 11;                  // / SEQ
      const int l   = m & (SEQ - 1);
      const int mat = n >> 10;                  // 0=q 1=k 2=v
      const int r   = n & (DIMM - 1);
      const int h   = r >> 6;
      const int d   = r & (HD - 1);
      const int bh  = b * NHEAD + h;
      if (mat == 0)      q_buf [(bh * SEQ + l) * HD + d] = v;
      else if (mat == 1) k_buf [(bh * SEQ + l) * HD + d] = v;
      else               vt_buf[(bh * HD + d) * SEQ + l] = v;
    }
}

// ---------------------------------------------------------------------------
// Kernel 2: flash attention. One wave per 16-query block, keys in blocks of 32.
// grid = (SEQ/(16*8), BATCH*NHEAD), block = 256 (8 waves).
// ---------------------------------------------------------------------------
__global__ __launch_bounds__(256) void attn_kernel(const __bf16* __restrict__ q_buf,
                                                   const __bf16* __restrict__ k_buf,
                                                   const __bf16* __restrict__ vt_buf,
                                                   __bf16* __restrict__ attn_buf) {
  __shared__ __bf16 pstage[8 * 16 * 32];  // 1KB per wave: P tile 16x32 row-major
  const int lane = threadIdx.x & 31;
  const int wid  = threadIdx.x >> 5;
  const int half = lane >> 4;
  const int bh = blockIdx.y;                      // 0..31
  const int q0 = (blockIdx.x * 8 + wid) * 16;

  const __bf16* Q  = q_buf  + (size_t)bh * SEQ * HD;
  const __bf16* K  = k_buf  + (size_t)bh * SEQ * HD;
  const __bf16* Vt = vt_buf + (size_t)bh * HD * SEQ;
  __bf16* pw = pstage + wid * 512;

  // Q fragments kept resident (16 queries x 64 dims = two 16x32 A-frags)
  const v16bf qa0 = load_frag_a(Q, HD, q0, 0);
  const v16bf qa1 = load_frag_a(Q, HD, q0, 32);

  v8f o[4] = {};
  float mrow[8], lrow[8];
#pragma unroll
  for (int j = 0; j < 8; ++j) { mrow[j] = -__builtin_inff(); lrow[j] = 0.0f; }

  for (int kb = 0; kb < SEQ; kb += 32) {
    if (kb + 32 < SEQ) {
      __builtin_prefetch(K  + (kb + 32) * HD, 0, 0);
      __builtin_prefetch(Vt + (kb + 32), 0, 0);
    }

    // S = Q K^T for 32 keys -> two 16x16 f32 tiles
    v8f s0 = {}, s1 = {};
    {
      v16bf b00 = load_frag_bt(K, HD, kb, 0);
      v16bf b01 = load_frag_bt(K, HD, kb, 32);
      v16bf b10 = load_frag_bt(K, HD, kb + 16, 0);
      v16bf b11 = load_frag_bt(K, HD, kb + 16, 32);
      s0 = WMMA_BF16(qa0, b00, s0);
      s0 = WMMA_BF16(qa1, b01, s0);
      s1 = WMMA_BF16(qa0, b10, s1);
      s1 = WMMA_BF16(qa1, b11, s1);
    }

    // online softmax: C-layout row r = j + 8*half lives across the 16 lanes of
    // this half-wave -> xor-shuffle reduce with masks 1,2,4,8.
    float corr[8];
#pragma unroll
    for (int j = 0; j < 8; ++j) {
      const float a = s0[j] * QK_SCALE;
      const float b = s1[j] * QK_SCALE;
      float v = fmaxf(a, b);
      v = fmaxf(v, __shfl_xor(v, 1, 32));
      v = fmaxf(v, __shfl_xor(v, 2, 32));
      v = fmaxf(v, __shfl_xor(v, 4, 32));
      v = fmaxf(v, __shfl_xor(v, 8, 32));
      const float mnew = fmaxf(mrow[j], v);
      corr[j] = __expf(mrow[j] - mnew);
      mrow[j] = mnew;
      const float p0 = __expf(a - mnew);
      const float p1 = __expf(b - mnew);
      float rs = p0 + p1;
      rs += __shfl_xor(rs, 1, 32);
      rs += __shfl_xor(rs, 2, 32);
      rs += __shfl_xor(rs, 4, 32);
      rs += __shfl_xor(rs, 8, 32);
      lrow[j] = lrow[j] * corr[j] + rs;
      // stage P (bf16) row-major 16x32 for the C->A relayout
      const int row = j + half * 8;
      pw[row * 32 + (lane & 15)]      = (__bf16)p0;
      pw[row * 32 + 16 + (lane & 15)] = (__bf16)p1;
    }
    // LDS ops of a wave are in-order (DScnt); just fence the compiler + wait.
    asm volatile("s_wait_dscnt 0x0" ::: "memory");
    const v16bf pa = load_frag_a(pw, 32, 0, 0);

    // O = diag(corr) * O + P * V   (V consumed via its transpose: contiguous B)
#pragma unroll
    for (int t = 0; t < 4; ++t) {
#pragma unroll
      for (int j = 0; j < 8; ++j) o[t][j] *= corr[j];
      v16bf bv = load_frag_bt(Vt, SEQ, t * 16, kb);
      o[t] = WMMA_BF16(pa, bv, o[t]);
    }
    asm volatile("" ::: "memory");  // keep next iteration's stores after loads
  }

#pragma unroll
  for (int j = 0; j < 8; ++j) lrow[j] = 1.0f / lrow[j];

  const int h = bh & (NHEAD - 1);
  const int b = bh >> 4;
#pragma unroll
  for (int t = 0; t < 4; ++t)
#pragma unroll
    for (int j = 0; j < 8; ++j) {
      const int row = b * SEQ + q0 + j + half * 8;
      const int col = h * HD + t * 16 + (lane & 15);
      attn_buf[row * DIMM + col] = (__bf16)(o[t][j] * lrow[j]);
    }
}

// ---------------------------------------------------------------------------
// Kernel 3: out = attn @ w_out^T  (M=4096, N=1024, K=1024) -> fp32 d_out
// ---------------------------------------------------------------------------
__global__ __launch_bounds__(256) void out_gemm(const __bf16* __restrict__ A,
                                                const __bf16* __restrict__ W,
                                                float* __restrict__ out) {
  const int lane = threadIdx.x & 31;
  const int wid  = threadIdx.x >> 5;
  const int half = lane >> 4;
  const int m0 = blockIdx.x * 16;
  const int n0 = blockIdx.y * 512 + wid * 64;

  v8f acc[4] = {};
  gemm_tile_16x64(A, W, DIMM, m0, n0, acc);

#pragma unroll
  for (int t = 0; t < 4; ++t)
#pragma unroll
    for (int j = 0; j < 8; ++j) {
      const int m = m0 + j + half * 8;
      const int n = n0 + t * 16 + (lane & 15);
      out[m * DIMM + n] = acc[t][j];
    }
}

// ---------------------------------------------------------------------------
extern "C" void kernel_launch(void* const* d_in, const int* in_sizes, int n_in,
                              void* d_out, int out_size, void* d_ws, size_t ws_size,
                              hipStream_t stream) {
  const float* x     = (const float*)d_in[0];  // [2,2048,1024]
  const float* w_qkv = (const float*)d_in[1];  // [3072,1024]
  const float* w_out = (const float*)d_in[2];  // [1024,1024]
  float* out = (float*)d_out;

  char* ws = (char*)d_ws;
  __bf16* xbf   = (__bf16*)(ws);                      //  8 MiB
  __bf16* wqbf  = (__bf16*)(ws + (size_t)(8  << 20)); //  6 MiB
  __bf16* wobf  = (__bf16*)(ws + (size_t)(14 << 20)); //  2 MiB
  __bf16* qb    = (__bf16*)(ws + (size_t)(16 << 20)); //  8 MiB  [B,H,L,hd]
  __bf16* kbuf  = (__bf16*)(ws + (size_t)(24 << 20)); //  8 MiB  [B,H,L,hd]
  __bf16* vtb   = (__bf16*)(ws + (size_t)(32 << 20)); //  8 MiB  [B,H,hd,L]
  __bf16* abuf  = (__bf16*)(ws + (size_t)(40 << 20)); //  8 MiB  [B*L, D]

  cvt_f32_bf16<<<512, 256, 0, stream>>>(x,     xbf,  NTOK * DIMM);
  cvt_f32_bf16<<<512, 256, 0, stream>>>(w_qkv, wqbf, 3 * DIMM * DIMM);
  cvt_f32_bf16<<<512, 256, 0, stream>>>(w_out, wobf, DIMM * DIMM);

  qkv_gemm<<<dim3(NTOK / 16, 6), 256, 0, stream>>>(xbf, wqbf, qb, kbuf, vtb);
  attn_kernel<<<dim3(SEQ / 128, BATCH * NHEAD), 256, 0, stream>>>(qb, kbuf, vtb, abuf);
  out_gemm<<<dim3(NTOK / 16, 2), 256, 0, stream>>>(abuf, wobf, out);
}
